// AttentionBlock_78993038508293
// MI455X (gfx1250) — compile-verified, matrix operand order checked by reading, not run on hardware
//
#include <hip/hip_runtime.h>
#include <hip/hip_fp16.h>

// MI455X / gfx1250, wave32. f16 WMMA flash attention (f32 accumulate).
// B=2, S=2048, D=1024, H=16, HD=64.

typedef __attribute__((ext_vector_type(16))) _Float16 v16h;
typedef __attribute__((ext_vector_type(8)))  float    v8f;

#define S_LEN  2048
#define HD     64
#define NHEAD  16
#define BATCH  2
#define DMODEL 1024

union FragH {
    uint4 q[2];
    unsigned int d[8];
    unsigned short u[16];
    v16h v;
};
union AccF {
    v8f v;
    float f[8];
};

// Pack two f32 -> packed 2 x f16 (low = a, high = b) in one VALU op.
__device__ __forceinline__ unsigned int pack_h2(float a, float b) {
#if __has_builtin(__builtin_amdgcn_cvt_pkrtz)
    auto r = __builtin_amdgcn_cvt_pkrtz(a, b);   // v_cvt_pk_rtz_f16_f32
    return __builtin_bit_cast(unsigned int, r);
#else
    union { _Float16 h[2]; unsigned int u; } x;
    x.h[0] = (_Float16)a; x.h[1] = (_Float16)b;
    return x.u;
#endif
}

// 16x32 f16 fragment from a row-major f16 matrix.
// A-frag: row = M (lane%16), cols = K.  B-frag: row = N (lane%16), cols = K.
// Per-lane data: K in [half*8, half*8+8) -> vgpr0-3, [16+half*8, +8) -> vgpr4-7.
__device__ __forceinline__ v16h load_frag_h(const unsigned short* __restrict__ base,
                                            int row, int stride, int colBase, int lane) {
    const int half = lane >> 4;
    const unsigned short* p = base + (size_t)row * stride + colBase + half * 8;
    FragH f;
    f.q[0] = *(const uint4*)(p);
    f.q[1] = *(const uint4*)(p + 16);
    return f.v;
}

// Same fragment, but source is f32 row-major; packed-convert on the fly.
__device__ __forceinline__ v16h load_frag_f32(const float* __restrict__ base,
                                              int row, int stride, int colBase, int lane) {
    const int half = lane >> 4;
    const float* p = base + (size_t)row * stride + colBase + half * 8;
    float4 a0 = *(const float4*)(p);
    float4 a1 = *(const float4*)(p + 4);
    float4 b0 = *(const float4*)(p + 16);
    float4 b1 = *(const float4*)(p + 20);
    FragH f;
    f.d[0] = pack_h2(a0.x, a0.y);  f.d[1] = pack_h2(a0.z, a0.w);
    f.d[2] = pack_h2(a1.x, a1.y);  f.d[3] = pack_h2(a1.z, a1.w);
    f.d[4] = pack_h2(b0.x, b0.y);  f.d[5] = pack_h2(b0.z, b0.w);
    f.d[6] = pack_h2(b1.x, b1.y);  f.d[7] = pack_h2(b1.z, b1.w);
    return f.v;
}

// XOR butterfly within the 32 lanes (group-of-32 ds_swizzle; masks 1,2,4,8
// reduce fully inside each 16-lane half -> per-row reduction for C-layout).
template <int MASK>
__device__ __forceinline__ float xor_bfly(float v) {
    int i = __builtin_bit_cast(int, v);
    i = __builtin_amdgcn_ds_swizzle(i, (MASK << 10) | 0x1f);
    return __builtin_bit_cast(float, i);
}

template <int MASK>
__device__ __forceinline__ float red_max(float v) { return fmaxf(v, xor_bfly<MASK>(v)); }
template <int MASK>
__device__ __forceinline__ float red_sum(float v) { return v + xor_bfly<MASK>(v); }

// ---------------------------------------------------------------------------
// Kernel 1: QKV projection. xh[b,h] is the contiguous (S, HD) block at
// x + bh*S*HD (raw reshape semantics). Each wave: one 16-row s-tile of one bh.
// Q gets the softmax scale (1/8) folded in. V is stored transposed: Vt[e][s].
// ---------------------------------------------------------------------------
__global__ void __launch_bounds__(128)
qkv_proj_kernel(const float* __restrict__ x,
                const float* __restrict__ Wq, const float* __restrict__ bq,
                const float* __restrict__ Wk, const float* __restrict__ bk,
                const float* __restrict__ Wv, const float* __restrict__ bv,
                unsigned short* __restrict__ Qw,
                unsigned short* __restrict__ Kw,
                unsigned short* __restrict__ Vt) {
    const int lane  = threadIdx.x & 31;
    const int wave  = threadIdx.x >> 5;
    const int w     = blockIdx.x * 4 + wave;
    const int sTile = w & (S_LEN / 16 - 1);     // 128 tiles per bh
    const int bh    = w >> 7;                    // 0..31
    const int h     = bh & (NHEAD - 1);
    const int sBase = sTile * 16;
    const int half  = lane >> 4;
    const int ln    = lane & 15;

    const float* xh = x + (size_t)bh * S_LEN * HD;
    v16h Xa0 = load_frag_f32(xh, sBase + ln, HD, 0,  lane);
    v16h Xa1 = load_frag_f32(xh, sBase + ln, HD, 32, lane);

    const float* Wm[3] = { Wq + (size_t)h * HD * HD,
                           Wk + (size_t)h * HD * HD,
                           Wv + (size_t)h * HD * HD };
    const float* bm[3] = { bq + h * HD, bk + h * HD, bv + h * HD };

#pragma unroll
    for (int mtx = 0; mtx < 3; ++mtx) {
#pragma unroll
        for (int t = 0; t < 4; ++t) {
            const int d = t * 16 + ln;                 // output column (N)
            const float bias = bm[mtx][d];
            AccF acc;
#pragma unroll
            for (int r = 0; r < 8; ++r) acc.f[r] = bias;
            v16h Wb0 = load_frag_f32(Wm[mtx], d, HD, 0,  lane);
            v16h Wb1 = load_frag_f32(Wm[mtx], d, HD, 32, lane);
            acc.v = __builtin_amdgcn_wmma_f32_16x16x32_f16(false, Xa0, false, Wb0,
                                                           (short)0, acc.v, false, false);
            acc.v = __builtin_amdgcn_wmma_f32_16x16x32_f16(false, Xa1, false, Wb1,
                                                           (short)0, acc.v, false, false);
            if (mtx == 0) {
#pragma unroll
                for (int r = 0; r < 8; ++r) acc.f[r] *= 0.125f;   // 1/sqrt(HD)
            }
            if (mtx == 2) {
                unsigned short* vt = Vt + (size_t)bh * HD * S_LEN;
#pragma unroll
                for (int r = 0; r < 8; r += 2) {
                    const unsigned int pk = pack_h2(acc.f[r], acc.f[r + 1]);
                    const int s0 = sBase + r + 8 * half;
                    vt[(size_t)d * S_LEN + s0]     = (unsigned short)pk;
                    vt[(size_t)d * S_LEN + s0 + 1] = (unsigned short)(pk >> 16);
                }
            } else {
                unsigned short* dst = (mtx == 0 ? Qw : Kw) + (size_t)bh * S_LEN * HD;
#pragma unroll
                for (int r = 0; r < 8; r += 2) {
                    const unsigned int pk = pack_h2(acc.f[r], acc.f[r + 1]);
                    const int s0 = sBase + r + 8 * half;
                    dst[(size_t)s0 * HD + d]       = (unsigned short)pk;
                    dst[(size_t)(s0 + 1) * HD + d] = (unsigned short)(pk >> 16);
                }
            }
        }
    }
}

// ---------------------------------------------------------------------------
// Kernel 2: flash attention. One wave per 16-query tile; online softmax over
// all 2048 keys in 64-key steps (4 S-tiles -> halves per-key reduction cost).
// K/V f16 tiles stream from the 192MB L2. P relaid C-layout -> A-layout via
// a per-wave LDS tile (stride 72 halfwords: 16B-aligned, conflict-free b128).
// ---------------------------------------------------------------------------
__global__ void __launch_bounds__(128)
flash_attn_kernel(const unsigned short* __restrict__ Qw,
                  const unsigned short* __restrict__ Kw,
                  const unsigned short* __restrict__ Vt,
                  float* __restrict__ out) {
    __shared__ unsigned short plds_all[4][16 * 72];
    const int lane  = threadIdx.x & 31;
    const int wave  = threadIdx.x >> 5;
    const int w     = blockIdx.x * 4 + wave;
    const int qTile = w & (S_LEN / 16 - 1);
    const int bh    = w >> 7;
    const int b     = bh >> 4;
    const int h     = bh & 15;
    const int sBase = qTile * 16;
    const int half  = lane >> 4;
    const int ln    = lane & 15;
    unsigned short* plds = plds_all[wave];

    const unsigned short* Qb = Qw + (size_t)bh * S_LEN * HD;
    const unsigned short* Kb = Kw + (size_t)bh * S_LEN * HD;
    const unsigned short* Vb = Vt + (size_t)bh * HD * S_LEN;

    v16h Qa0 = load_frag_h(Qb, sBase + ln, HD, 0,  lane);
    v16h Qa1 = load_frag_h(Qb, sBase + ln, HD, 32, lane);

    AccF o[4];
#pragma unroll
    for (int t = 0; t < 4; ++t)
#pragma unroll
        for (int r = 0; r < 8; ++r) o[t].f[r] = 0.0f;
    float mrow[8], lrow[8];
#pragma unroll
    for (int r = 0; r < 8; ++r) { mrow[r] = -1e30f; lrow[r] = 0.0f; }

    const v8f zacc = {0.f, 0.f, 0.f, 0.f, 0.f, 0.f, 0.f, 0.f};

    for (int n0 = 0; n0 < S_LEN; n0 += 64) {
        // S = Q @ K^T for 64 keys (scale already folded into Q)
        AccF s[4];
#pragma unroll
        for (int j = 0; j < 4; ++j) {
            v16h kb0 = load_frag_h(Kb, n0 + 16 * j + ln, HD, 0,  lane);
            v16h kb1 = load_frag_h(Kb, n0 + 16 * j + ln, HD, 32, lane);
            s[j].v = __builtin_amdgcn_wmma_f32_16x16x32_f16(false, Qa0, false, kb0,
                                                            (short)0, zacc, false, false);
            s[j].v = __builtin_amdgcn_wmma_f32_16x16x32_f16(false, Qa1, false, kb1,
                                                            (short)0, s[j].v, false, false);
        }

        // Online softmax. C-layout: vgpr r, lane -> row M = r + 8*half, col N = ln.
#pragma unroll
        for (int r = 0; r < 8; ++r) {
            float vmax = fmaxf(fmaxf(s[0].f[r], s[1].f[r]), fmaxf(s[2].f[r], s[3].f[r]));
            vmax = red_max<1>(vmax); vmax = red_max<2>(vmax);
            vmax = red_max<4>(vmax); vmax = red_max<8>(vmax);
            const float mn    = fmaxf(mrow[r], vmax);
            const float alpha = __expf(mrow[r] - mn);
            mrow[r] = mn;
            const float p0 = __expf(s[0].f[r] - mn);
            const float p1 = __expf(s[1].f[r] - mn);
            const float p2 = __expf(s[2].f[r] - mn);
            const float p3 = __expf(s[3].f[r] - mn);
            float rs = (p0 + p1) + (p2 + p3);
            rs = red_sum<1>(rs); rs = red_sum<2>(rs);
            rs = red_sum<4>(rs); rs = red_sum<8>(rs);
            lrow[r] = lrow[r] * alpha + rs;
#pragma unroll
            for (int t = 0; t < 4; ++t) o[t].f[r] *= alpha;
            const int mloc = r + 8 * half;
            const unsigned int pkA = pack_h2(p0, p1);
            const unsigned int pkB = pack_h2(p2, p3);
            plds[mloc * 72 + ln]      = (unsigned short)pkA;
            plds[mloc * 72 + 16 + ln] = (unsigned short)(pkA >> 16);
            plds[mloc * 72 + 32 + ln] = (unsigned short)pkB;
            plds[mloc * 72 + 48 + ln] = (unsigned short)(pkB >> 16);
        }

        // Reload P as two A-fragments (16x32 each; M=query, K=local key).
        FragH pf0, pf1;
        {
            const unsigned short* pp = plds + ln * 72 + half * 8;
            pf0.q[0] = *(const uint4*)(pp);
            pf0.q[1] = *(const uint4*)(pp + 16);
            pf1.q[0] = *(const uint4*)(pp + 32);
            pf1.q[1] = *(const uint4*)(pp + 48);
        }

        // O += P @ V ; B-frags from Vt rows (row = e, contiguous key index).
#pragma unroll
        for (int t = 0; t < 4; ++t) {
            v16h vb0 = load_frag_h(Vb, t * 16 + ln, S_LEN, n0,      lane);
            v16h vb1 = load_frag_h(Vb, t * 16 + ln, S_LEN, n0 + 32, lane);
            o[t].v = __builtin_amdgcn_wmma_f32_16x16x32_f16(false, pf0.v, false, vb0,
                                                            (short)0, o[t].v, false, false);
            o[t].v = __builtin_amdgcn_wmma_f32_16x16x32_f16(false, pf1.v, false, vb1,
                                                            (short)0, o[t].v, false, false);
        }
    }

    float linv[8];
#pragma unroll
    for (int r = 0; r < 8; ++r) linv[r] = 1.0f / lrow[r];

    // final[b, s, h*HD + e] = O[b,h,s,e]
#pragma unroll
    for (int t = 0; t < 4; ++t) {
        const int e = t * 16 + ln;
#pragma unroll
        for (int r = 0; r < 8; ++r) {
            const int srow = sBase + r + 8 * half;
            out[(size_t)b * S_LEN * DMODEL + (size_t)srow * DMODEL + h * HD + e] =
                o[t].f[r] * linv[r];
        }
    }
}

extern "C" void kernel_launch(void* const* d_in, const int* in_sizes, int n_in,
                              void* d_out, int out_size, void* d_ws, size_t ws_size,
                              hipStream_t stream) {
    (void)in_sizes; (void)n_in; (void)out_size; (void)ws_size;
    const float* x  = (const float*)d_in[0];
    const float* Wq = (const float*)d_in[1];
    const float* bq = (const float*)d_in[2];
    const float* Wk = (const float*)d_in[3];
    const float* bk = (const float*)d_in[4];
    const float* Wv = (const float*)d_in[5];
    const float* bv = (const float*)d_in[6];
    float* out = (float*)d_out;

    // Workspace: f16 Q, K (row-major [bh][s][e]) and V transposed ([bh][e][s]).
    const size_t per = (size_t)BATCH * NHEAD * S_LEN * HD;   // 4,194,304 elems
    unsigned short* Qw = (unsigned short*)d_ws;
    unsigned short* Kw = Qw + per;
    unsigned short* Vt = Kw + per;                           // total 24 MiB

    const int nwaves = BATCH * NHEAD * (S_LEN / 16);         // 4096 waves
    dim3 blk(128);                                           // 4 waves/block
    dim3 grid(nwaves / 4);                                   // 1024 blocks

    hipLaunchKernelGGL(qkv_proj_kernel, grid, blk, 0, stream,
                       x, Wq, bq, Wk, bk, Wv, bv, Qw, Kw, Vt);
    hipLaunchKernelGGL(flash_attn_kernel, grid, blk, 0, stream,
                       Qw, Kw, Vt, out);
}
